// ShootBoth_37091337568588
// MI455X (gfx1250) — compile-verified, matrix operand order checked by reading, not run on hardware
//
#include <hip/hip_runtime.h>

#define NVOX (1 << 21)   // 128^3

typedef __attribute__((ext_vector_type(16))) _Float16 v16h;
typedef __attribute__((ext_vector_type(8)))  float    v8f;

// ---------------------------------------------------------------------------
// Row-base helpers for DFT-along-axis passes over a 128^3 volume.
// axis 0 stride 16384 (rows enumerate x1,x2), axis 1 stride 128 (rows x0,x2),
// axis 2 stride 1 (rows x0,x1).  For axes 0/1 consecutive rows are consecutive
// addresses (x2 fastest), for axis 2 the K index is the contiguous one.
// ---------------------------------------------------------------------------
__device__ __forceinline__ int rowBase(int r, int axis) {
  if (axis == 0) return r;                            // r = x1*128 + x2
  if (axis == 1) return ((r >> 7) << 14) | (r & 127); // r = x0*128 + x2
  return r << 7;                                      // r = x0*128 + x1
}

// CDNA5 16-bit A/B fragment (16x32) K index for half h (0..15), lane group grp.
__device__ __forceinline__ int kidx(int h, int grp) {
  return ((h >> 3) << 4) | (((h >> 1) & 3) << 1) | (grp << 3) | (h & 1);
}

// Cooperative tile (row,k) mapping chosen per axis for coalesced global access.
template <int AXIS>
__device__ __forceinline__ void tileRC(int e, int& row, int& k) {
  if (AXIS == 2) { row = e >> 7; k = e & 127; }   // k contiguous in memory
  else           { row = e & 15; k = e >> 4; }    // rows contiguous in memory
}

// ---------------------------------------------------------------------------
// One complex DFT pass along AXIS: Y[r,k] = sum_n X[r,n] * (C + i*sSign*S)[k,n]
// Block = 128 threads (4 waves) handles one 16-row tile x all 128 outputs.
// X tile staged via LDS (f16); B fragments loaded from fragment-ordered
// tables; results routed back through LDS for coalesced stores.
// ---------------------------------------------------------------------------
template <int AXIS, bool HASIM>
__global__ void dft_pass_kernel(const float* __restrict__ xre,
                                const float* __restrict__ xim,
                                float* __restrict__ yre,
                                float* __restrict__ yim,
                                const _Float16* __restrict__ cosF,
                                const _Float16* __restrict__ sinF,
                                float sSign, float scale) {
  constexpr int STRIDE = (AXIS == 0) ? 16384 : ((AXIS == 1) ? 128 : 1);
  __shared__ _Float16 sRe[16 * 128];
  __shared__ _Float16 sIm[16 * 128];
  const int t = threadIdx.x;          // 0..127
  const int rowTile = blockIdx.x;     // 0..1023

  // ---- cooperative coalesced load of the 16x128 tile -> LDS (f16) ----
  {
    float tr[16], ti[16];
    int li[16];
#pragma unroll
    for (int it = 0; it < 16; ++it) {
      const int e = it * 128 + t;
      int row, k; tileRC<AXIS>(e, row, k);
      const int off = rowBase(rowTile * 16 + row, AXIS) + k * STRIDE;
      tr[it] = xre[off];
      if (HASIM) ti[it] = xim[off];
      li[it] = row * 128 + k;
    }
#pragma unroll
    for (int it = 0; it < 16; ++it) {
      sRe[li[it]] = (_Float16)tr[it];
      if (HASIM) sIm[li[it]] = (_Float16)ti[it];
    }
  }
  __syncthreads();

  const int lane = t & 31;
  const int wid  = t >> 5;            // wave 0..3 -> col tiles 2*wid, 2*wid+1
  const int grp  = lane >> 4;
  const int l15  = lane & 15;
  const unsigned* sRe32 = (const unsigned*)sRe;
  const unsigned* sIm32 = (const unsigned*)sIm;

  v8f aRC[2] = {}; v8f aRS[2] = {}; v8f aIS[2] = {}; v8f aIC[2] = {};
#pragma unroll
  for (int kt = 0; kt < 4; ++kt) {
    // A fragments from LDS: 8 aligned dword reads each (K pairs are adjacent).
    union { v16h v; unsigned u[8]; } fRe, fIm;
#pragma unroll
    for (int i = 0; i < 8; ++i) {
      const int kb = kt * 32 + ((i < 4) ? (2 * i + (grp << 3))
                                        : (16 + 2 * (i - 4) + (grp << 3)));
      const int du = ((l15 << 7) + kb) >> 1;
      fRe.u[i] = sRe32[du];
      if (HASIM) fIm.u[i] = sIm32[du];
    }
#pragma unroll
    for (int ct = 0; ct < 2; ++ct) {
      const int fb = ((((wid << 1) | ct) * 4 + kt) * 32 + lane) * 16;
      const v16h bC = *(const v16h*)(cosF + fb);   // contiguous 32B per lane
      const v16h bS = *(const v16h*)(sinF + fb);
      aRC[ct] = __builtin_amdgcn_wmma_f32_16x16x32_f16(false, fRe.v, false, bC, (short)0, aRC[ct], false, false);
      aRS[ct] = __builtin_amdgcn_wmma_f32_16x16x32_f16(false, fRe.v, false, bS, (short)0, aRS[ct], false, false);
      if (HASIM) {
        aIS[ct] = __builtin_amdgcn_wmma_f32_16x16x32_f16(false, fIm.v, false, bS, (short)0, aIS[ct], false, false);
        aIC[ct] = __builtin_amdgcn_wmma_f32_16x16x32_f16(false, fIm.v, false, bC, (short)0, aIC[ct], false, false);
      }
    }
  }
  __syncthreads();   // all waves done reading the A tile

  // ---- combine and park results in LDS (f16) ----
#pragma unroll
  for (int ct = 0; ct < 2; ++ct) {
    const int col = ((wid << 1) | ct) * 16 + l15;
#pragma unroll
    for (int j = 0; j < 8; ++j) {
      const int row = j + (grp << 3);
      float re, im;
      if (HASIM) {
        re = scale * (aRC[ct][j] - sSign * aIS[ct][j]);
        im = scale * (sSign * aRS[ct][j] + aIC[ct][j]);
      } else {
        re = scale * aRC[ct][j];
        im = scale * sSign * aRS[ct][j];
      }
      sRe[row * 128 + col] = (_Float16)re;
      sIm[row * 128 + col] = (_Float16)im;
    }
  }
  __syncthreads();

  // ---- cooperative coalesced store ----
#pragma unroll
  for (int it = 0; it < 16; ++it) {
    const int e = it * 128 + t;
    int row, k; tileRC<AXIS>(e, row, k);
    const int off = rowBase(rowTile * 16 + row, AXIS) + k * STRIDE;
    yre[off] = (float)sRe[row * 128 + k];
    yim[off] = (float)sIm[row * 128 + k];
  }
}

// ---------------------------------------------------------------------------
// Frequency-domain metric multiply: kern = 1e-3 + lam + 0.1*lam^2,
// lam = sum_axes (2 - 2cos(2*pi*x/128)).  inverse=1 -> multiply by 1/kern.
// ---------------------------------------------------------------------------
__global__ void freq_scale_kernel(float* __restrict__ re, float* __restrict__ im,
                                  int inverse) {
  const int i  = blockIdx.x * blockDim.x + threadIdx.x;
  const int x0 = i >> 14, x1 = (i >> 7) & 127, x2 = i & 127;
  const float c = 0.049087385212340517f;  // 2*pi/128
  const float lam = 6.0f - 2.0f * (cosf(c * (float)x0) + cosf(c * (float)x1) + cosf(c * (float)x2));
  const float kern = 1.0e-3f + lam + 0.1f * lam * lam;
  const float fac = inverse ? (1.0f / kern) : kern;
  re[i] *= fac;
  im[i] *= fac;
}

// DFT factor tables written directly in per-lane WMMA B-fragment order:
// idx = ((colTile*4 + kt)*32 + lane)*16 + h.
__global__ void init_tables_kernel(_Float16* __restrict__ cosF,
                                   _Float16* __restrict__ sinF) {
  const int i    = blockIdx.x * blockDim.x + threadIdx.x;  // 16384
  const int h    = i & 15;
  const int lane = (i >> 4) & 31;
  const int kt   = (i >> 9) & 3;
  const int ct   = i >> 11;
  const int grp  = lane >> 4;
  const int ncol = ct * 16 + (lane & 15);
  const int k    = kt * 32 + kidx(h, grp);
  const float ang = 0.049087385212340517f * (float)((ncol * k) & 127);
  cosF[i] = (_Float16)cosf(ang);
  sinF[i] = (_Float16)sinf(ang);
}

// ---------------------------------------------------------------------------
// Trilinear sample with circular wrap of a 3-channel (channel-major) field.
// ---------------------------------------------------------------------------
__device__ __forceinline__ void sample3(const float* __restrict__ f,
                                        float cx, float cy, float cz, float* o) {
  const float fx = floorf(cx), fy = floorf(cy), fz = floorf(cz);
  const float tx = cx - fx, ty = cy - fy, tz = cz - fz;
  const int x0 = ((int)fx) & 127, x1 = (x0 + 1) & 127;
  const int y0 = ((int)fy) & 127, y1 = (y0 + 1) & 127;
  const int z0 = ((int)fz) & 127, z1 = (z0 + 1) & 127;
  const int i000 = (x0 << 14) | (y0 << 7) | z0, i001 = (x0 << 14) | (y0 << 7) | z1;
  const int i010 = (x0 << 14) | (y1 << 7) | z0, i011 = (x0 << 14) | (y1 << 7) | z1;
  const int i100 = (x1 << 14) | (y0 << 7) | z0, i101 = (x1 << 14) | (y0 << 7) | z1;
  const int i110 = (x1 << 14) | (y1 << 7) | z0, i111 = (x1 << 14) | (y1 << 7) | z1;
  const float w000 = (1.f - tx) * (1.f - ty) * (1.f - tz), w001 = (1.f - tx) * (1.f - ty) * tz;
  const float w010 = (1.f - tx) * ty * (1.f - tz),         w011 = (1.f - tx) * ty * tz;
  const float w100 = tx * (1.f - ty) * (1.f - tz),         w101 = tx * (1.f - ty) * tz;
  const float w110 = tx * ty * (1.f - tz),                 w111 = tx * ty * tz;
#pragma unroll
  for (int c = 0; c < 3; ++c) {
    const float* g = f + (size_t)c * NVOX;
    o[c] = w000 * g[i000] + w001 * g[i001] + w010 * g[i010] + w011 * g[i011]
         + w100 * g[i100] + w101 * g[i101] + w110 * g[i110] + w111 * g[i111];
  }
}

// phi_new(x) = phi(x) + dt * v(x + phi(x))
__global__ void update_phi_kernel(const float* __restrict__ phiS,
                                  const float* __restrict__ v,
                                  float* __restrict__ phiD, float dt) {
  const int i  = blockIdx.x * blockDim.x + threadIdx.x;
  const int x0 = i >> 14, x1 = (i >> 7) & 127, x2 = i & 127;
  const float d0 = phiS[i], d1 = phiS[i + NVOX], d2 = phiS[i + 2 * NVOX];
  float s[3];
  sample3(v, (float)x0 + d0, (float)x1 + d1, (float)x2 + d2, s);
  phiD[i]            = d0 + dt * s[0];
  phiD[i + NVOX]     = d1 + dt * s[1];
  phiD[i + 2 * NVOX] = d2 + dt * s[2];
}

// psi_new(x) = psi(x - dt*v(x)) - dt*v(x)
__global__ void update_psi_kernel(const float* __restrict__ psiS,
                                  const float* __restrict__ v,
                                  float* __restrict__ psiD, float dt) {
  const int i  = blockIdx.x * blockDim.x + threadIdx.x;
  const int x0 = i >> 14, x1 = (i >> 7) & 127, x2 = i & 127;
  const float v0 = v[i], v1 = v[i + NVOX], v2 = v[i + 2 * NVOX];
  float s[3];
  sample3(psiS, (float)x0 - dt * v0, (float)x1 - dt * v1, (float)x2 - dt * v2, s);
  psiD[i]            = s[0] - dt * v0;
  psiD[i + NVOX]     = s[1] - dt * v1;
  psiD[i + 2 * NVOX] = s[2] - dt * v2;
}

// m_b = det(J) * sum_a J[a][b] * m0(x + psi(x))_a,  J[a][b] = central diff + I
__global__ void momentum_kernel(const float* __restrict__ psi,
                                const float* __restrict__ m0,
                                float* __restrict__ m) {
  const int i  = blockIdx.x * blockDim.x + threadIdx.x;
  const int x0 = i >> 14, x1 = (i >> 7) & 127, x2 = i & 127;
  const int ip[3]  = { ((((x0 + 1) & 127)) << 14) | (x1 << 7) | x2,
                       (x0 << 14) | ((((x1 + 1) & 127)) << 7) | x2,
                       (x0 << 14) | (x1 << 7) | ((x2 + 1) & 127) };
  const int in_[3] = { ((((x0 + 127) & 127)) << 14) | (x1 << 7) | x2,
                       (x0 << 14) | ((((x1 + 127) & 127)) << 7) | x2,
                       (x0 << 14) | (x1 << 7) | ((x2 + 127) & 127) };
  float J[3][3];
#pragma unroll
  for (int a = 0; a < 3; ++a) {
#pragma unroll
    for (int b = 0; b < 3; ++b) {
      const float* ub = psi + (size_t)b * NVOX;
      J[a][b] = 0.5f * (ub[ip[a]] - ub[in_[a]]) + ((a == b) ? 1.0f : 0.0f);
    }
  }
  const float det = J[0][0] * (J[1][1] * J[2][2] - J[1][2] * J[2][1])
                  - J[0][1] * (J[1][0] * J[2][2] - J[1][2] * J[2][0])
                  + J[0][2] * (J[1][0] * J[2][1] - J[1][1] * J[2][0]);
  const float d0 = psi[i], d1 = psi[i + NVOX], d2 = psi[i + 2 * NVOX];
  float s[3];
  sample3(m0, (float)x0 + d0, (float)x1 + d1, (float)x2 + d2, s);
#pragma unroll
  for (int b = 0; b < 3; ++b)
    m[(size_t)b * NVOX + i] = det * (J[0][b] * s[0] + J[1][b] * s[1] + J[2][b] * s[2]);
}

__global__ void zero_kernel(float* __restrict__ p, int n) {
  const int i = blockIdx.x * blockDim.x + threadIdx.x;
  if (i < n) p[i] = 0.0f;
}
__global__ void copy_kernel(float* __restrict__ d, const float* __restrict__ s, int n) {
  const int i = blockIdx.x * blockDim.x + threadIdx.x;
  if (i < n) d[i] = s[i];
}

// ---------------------------------------------------------------------------
extern "C" void kernel_launch(void* const* d_in, const int* in_sizes, int n_in,
                              void* d_out, int out_size, void* d_ws, size_t ws_size,
                              hipStream_t stream) {
  (void)in_sizes; (void)n_in; (void)out_size; (void)ws_size;
  const float* vin = (const float*)d_in[0];   // [1,3,128,128,128] f32 (channel-major)
  float* out = (float*)d_out;                 // phi (3*N) then iphi (3*N)

  const size_t C  = (size_t)NVOX;
  const size_t CB = C * sizeof(float);
  const size_t FB = 3 * CB;
  char* w = (char*)d_ws;
  float* vbuf   = (float*)(w);
  float* m0buf  = (float*)(w + FB);
  float* phiAlt = (float*)(w + 2 * FB);
  float* psiAlt = (float*)(w + 3 * FB);
  float* mbuf   = (float*)(w + 4 * FB);
  float* reA    = (float*)(w + 5 * FB);
  float* imA    = (float*)(w + 5 * FB + CB);
  float* reB    = (float*)(w + 5 * FB + 2 * CB);
  float* imB    = (float*)(w + 5 * FB + 3 * CB);
  _Float16* cosF = (_Float16*)(w + 5 * FB + 4 * CB);
  _Float16* sinF = cosF + 16384;

  const float dt = 0.125f;
  float* phi0 = out;
  float* psi0 = out + 3 * C;

  copy_kernel<<<24576, 256, 0, stream>>>(vbuf, vin, 3 * NVOX);
  zero_kernel<<<24576, 256, 0, stream>>>(phi0, 3 * NVOX);
  zero_kernel<<<24576, 256, 0, stream>>>(psi0, 3 * NVOX);
  init_tables_kernel<<<64, 256, 0, stream>>>(cosF, sinF);

  // apply_kernel: normalized forward 3D DFT (1/128 per axis keeps spectra in
  // f16 range), multiply by kern (inverse=0) or 1/kern (=1), inverse 3D DFT
  // (conjugate sign, unit scale), real part -> y.
  auto applyK = [&](const float* x, float* y, int inverse) {
    const float fs = 0.0078125f;  // 1/128 per forward axis
    dft_pass_kernel<0, false><<<1024, 128, 0, stream>>>(x,   nullptr, reB, imB, cosF, sinF, -1.0f, fs);
    dft_pass_kernel<1, true ><<<1024, 128, 0, stream>>>(reB, imB,     reA, imA, cosF, sinF, -1.0f, fs);
    dft_pass_kernel<2, true ><<<1024, 128, 0, stream>>>(reA, imA,     reB, imB, cosF, sinF, -1.0f, fs);
    freq_scale_kernel<<<8192, 256, 0, stream>>>(reB, imB, inverse);
    dft_pass_kernel<0, true ><<<1024, 128, 0, stream>>>(reB, imB, reA, imA, cosF, sinF, 1.0f, 1.0f);
    dft_pass_kernel<1, true ><<<1024, 128, 0, stream>>>(reA, imA, reB, imB, cosF, sinF, 1.0f, 1.0f);
    dft_pass_kernel<2, true ><<<1024, 128, 0, stream>>>(reB, imB, y,   imA, cosF, sinF, 1.0f, 1.0f);
  };

  // m0 = L v0
  for (int c = 0; c < 3; ++c) applyK(vbuf + c * C, m0buf + c * C, 0);

  // 8 steps; even count => ping-pong ends back in d_out halves.
  float* pA = phi0; float* pB = phiAlt;
  float* qA = psi0; float* qB = psiAlt;
  for (int step = 0; step < 8; ++step) {
    update_phi_kernel<<<8192, 256, 0, stream>>>(pA, vbuf, pB, dt);
    update_psi_kernel<<<8192, 256, 0, stream>>>(qA, vbuf, qB, dt);
    momentum_kernel<<<8192, 256, 0, stream>>>(qB, m0buf, mbuf);
    for (int c = 0; c < 3; ++c) applyK(mbuf + c * C, vbuf + c * C, 1);  // v = K m
    float* t = pA; pA = pB; pB = t;
    t = qA; qA = qB; qB = t;
  }
}
